// CrystalGPT2_69810398429671
// MI455X (gfx1250) — matrix operand order repaired and checked by reading, not compiled, not run on hardware
//
#include <hip/hip_runtime.h>
#include <hip/hip_bf16.h>
#include <math.h>

// ---------------------------------------------------------------------------
// CrystalGPT2 forward on gfx1250 (MI455X).
// GEMMs: double-buffered LDS staging via global_load_async_to_lds_b128
// (ASYNCcnt) feeding ds_load_b128 fragment reads into v_wmma_f32_16x16x32_bf16.
// fp32 residual stream, bf16 operands everywhere else.
// ---------------------------------------------------------------------------

typedef __bf16 bf16;
typedef __attribute__((ext_vector_type(16))) bf16  v16bf;
typedef __attribute__((ext_vector_type(8)))  bf16  v8bf;
typedef __attribute__((ext_vector_type(8)))  float v8f;

#define DEVINL __device__ __forceinline__

// model dims
#define VOCAB 50257
#define DMODEL 768
#define NNEUR 256
#define NLAYER 4
#define NBATCH 4
#define NTOK 1024
#define NHID 1024
#define NROWS (NBATCH * NTOK)   // 4096

enum { EPI_BF16 = 0, EPI_F32 = 1, EPI_GELU = 2, EPI_RES = 3, EPI_BF16_T = 4, EPI_DIST = 5 };

DEVINL int imind(int a, int b) { return a < b ? a : b; }

// Issue an async 32-byte (2 x b128) global->LDS copy. INST_OFFSET applies to
// both the LDS and global addresses, so one vdst/vaddr pair covers both.
DEVINL void async_cp32(unsigned ldsOff, const bf16* g) {
  asm volatile("global_load_async_to_lds_b128 %0, %1, off\n\t"
               "global_load_async_to_lds_b128 %0, %1, off offset:16"
               :: "v"(ldsOff), "v"(g)
               : "memory");
}

// ---------------------------------------------------------------------------
// Generic bf16 GEMM: C(MxN) = A(MxK) @ B(KxN), B given transposed (Bt[n*K+k]).
// 256 threads = 8 waves. Workgroup tile 128x128; wave tile 32x64 (2x4 WMMA
// tiles); K-step 32. A/B slabs double-buffered in LDS (4 x 8KB).
// ---------------------------------------------------------------------------
__global__ __launch_bounds__(256) void gemm_bf16_kernel(
    const bf16* __restrict__ A, const bf16* __restrict__ Bt,
    float* __restrict__ Cf, bf16* __restrict__ Cb,
    const float* __restrict__ bias,
    const float* __restrict__ aux0,   // per-row  ||h||^2 (DIST)
    const float* __restrict__ aux1,   // per-col  ||p||^2 (DIST)
    const float* __restrict__ aux2,   // per-col  inter_scale (DIST)
    int M, int N, int K, int mode) {
  // LDS: A0 | A1 | B0 | B1, each 128x32 bf16 = 8KB
  __shared__ __align__(16) bf16 lds[4 * 128 * 32];

  const int tid  = threadIdx.x;
  const int lane = tid & 31;
  const int wave = tid >> 5;
  const int wr = wave >> 1;            // 0..3 (M strip)
  const int wc = wave & 1;             // 0..1 (N strip)
  const int tileM0 = blockIdx.y * 128;
  const int tileN0 = blockIdx.x * 128;

  // --- async copy assignments: thread copies 32B of one row/col per stage
  const int crow  = tid >> 1;          // 0..127
  const int chalf = tid & 1;           // which 16-element half of the 32-K slab
  const int arow = imind(tileM0 + crow, M - 1);
  const int bcol = imind(tileN0 + crow, N - 1);
  const bf16* aSrc = A  + (size_t)arow * K + chalf * 16;
  const bf16* bSrc = Bt + (size_t)bcol * K + chalf * 16;
  const unsigned ldsBase = (unsigned)(size_t)(&lds[0]);  // LDS offset = low 32 bits
  const unsigned aDst = ldsBase + (unsigned)(crow * 64 + chalf * 32);
  const unsigned bDst = ldsBase + 16384u + (unsigned)(crow * 64 + chalf * 32);

  v8f acc[2][4];
#pragma unroll
  for (int i = 0; i < 2; i++)
#pragma unroll
    for (int j = 0; j < 4; j++) acc[i][j] = (v8f)0.0f;

  const int nsteps = K >> 5;           // K is a multiple of 32 everywhere here

  // prologue: stage 0 -> buffer 0
  async_cp32(aDst, aSrc);
  async_cp32(bDst, bSrc);

  for (int s = 0; s < nsteps; ++s) {
    const int cur = s & 1;
    if (s + 1 < nsteps) {
      const unsigned boff = (unsigned)((cur ^ 1) * 8192);
      async_cp32(aDst + boff, aSrc + (size_t)(s + 1) * 32);
      async_cp32(bDst + boff, bSrc + (size_t)(s + 1) * 32);
      asm volatile("s_wait_asynccnt 0x4" ::: "memory");  // stage s complete
    } else {
      asm volatile("s_wait_asynccnt 0x0" ::: "memory");
    }
    __syncthreads();

    const bf16* la = lds + cur * 4096;          // A buffer (elements)
    const bf16* lb = lds + 8192 + cur * 4096;   // B buffer

    // A fragments: lanes 0-15 row tile+m, K={0..7,16..23}; lanes 16-31 K={8..15,24..31}
    v16bf af[2];
#pragma unroll
    for (int i = 0; i < 2; i++) {
      const int r  = wr * 32 + i * 16 + (lane & 15);
      const int kh = (lane >> 4) * 8;
      v8bf lo = *(const v8bf*)(la + r * 32 + kh);
      v8bf hi = *(const v8bf*)(la + r * 32 + 16 + kh);
#pragma unroll
      for (int e = 0; e < 8; e++) { af[i][e] = lo[e]; af[i][e + 8] = hi[e]; }
    }
    // B fragments: lanes 0-15 col, K={0..7,16..23}; lanes 16-31 col, K={8..15,24..31}
    v16bf bf[4];
#pragma unroll
    for (int j = 0; j < 4; j++) {
      const int c  = wc * 64 + j * 16 + (lane & 15);
      const int kb = (lane >> 4) * 8;
      v8bf lo = *(const v8bf*)(lb + c * 32 + kb);
      v8bf hi = *(const v8bf*)(lb + c * 32 + 16 + kb);
#pragma unroll
      for (int e = 0; e < 8; e++) { bf[j][e] = lo[e]; bf[j][e + 8] = hi[e]; }
    }

#pragma unroll
    for (int j = 0; j < 4; j++) {
      acc[0][j] = __builtin_amdgcn_wmma_f32_16x16x32_bf16(false, af[0], false, bf[j],
                                                          (short)0, acc[0][j], false, false);
      acc[1][j] = __builtin_amdgcn_wmma_f32_16x16x32_bf16(false, af[1], false, bf[j],
                                                          (short)0, acc[1][j], false, false);
    }
    __syncthreads();   // all waves done reading buffer `cur` before it is refilled
  }

  // epilogue. C tile layout: lanes 0-15 VGPR g -> row tile+g; lanes 16-31 -> row tile+8+g
#pragma unroll
  for (int i = 0; i < 2; i++) {
#pragma unroll
    for (int j = 0; j < 4; j++) {
      const int mBase = tileM0 + wr * 32 + i * 16 + ((lane >> 4) << 3);
      const int col = tileN0 + wc * 64 + j * 16 + (lane & 15);
      if (col >= N) continue;
      const float bval = bias ? bias[col] : 0.0f;
#pragma unroll
      for (int g = 0; g < 8; g++) {
        const int m = mBase + g;
        if (m >= M) continue;
        const float v = acc[i][j][g];
        if (mode == EPI_BF16) {
          Cb[(size_t)m * N + col] = (bf16)(v + bval);
        } else if (mode == EPI_F32) {
          Cf[(size_t)m * N + col] = v + bval;
        } else if (mode == EPI_GELU) {
          const float t = v + bval;
          Cb[(size_t)m * N + col] = (bf16)(0.5f * t * (1.0f + erff(t * 0.70710678118f)));
        } else if (mode == EPI_RES) {
          Cf[(size_t)m * N + col] += v + bval;
        } else if (mode == EPI_BF16_T) {
          Cb[(size_t)col * M + m] = (bf16)(v + bval);
        } else {  // EPI_DIST: d2 = ||h||^2 + ||p||^2 - 2 h.p ; logit = scale/(dist+0.1)
          const float d2 = aux0[m] + aux1[col] - 2.0f * v;
          const float dist = sqrtf(fmaxf(d2, 0.0f));
          Cf[(size_t)m * N + col] = aux2[col] / (dist + 0.1f);
        }
      }
    }
  }
}

// ---------------------------------------------------------------------------
// LayerNorm: one wave per row of D=768. Writes bf16 output + optional sum(h^2).
// ---------------------------------------------------------------------------
__global__ __launch_bounds__(256) void ln_kernel(const float* __restrict__ X,
                                                 const float* __restrict__ w,
                                                 const float* __restrict__ b,
                                                 bf16* __restrict__ H,
                                                 float* __restrict__ HH,
                                                 int rows, int D) {
  const int lane = threadIdx.x & 31;
  const int wave = threadIdx.x >> 5;
  const int row = blockIdx.x * 8 + wave;
  if (row >= rows) return;
  const float* x = X + (size_t)row * D;
  float s = 0.0f, s2 = 0.0f;
  for (int i = lane; i < D; i += 32) { const float v = x[i]; s += v; s2 += v * v; }
#pragma unroll
  for (int off = 16; off > 0; off >>= 1) { s += __shfl_xor(s, off); s2 += __shfl_xor(s2, off); }
  const float mean = s / (float)D;
  const float var = s2 / (float)D - mean * mean;
  const float rstd = rsqrtf(var + 1e-5f);
  bf16* h = H + (size_t)row * D;
  float hh = 0.0f;
  for (int i = lane; i < D; i += 32) {
    const float v = (x[i] - mean) * rstd * w[i] + b[i];
    h[i] = (bf16)v;
    hh += v * v;
  }
  if (HH) {
#pragma unroll
    for (int off = 16; off > 0; off >>= 1) hh += __shfl_xor(hh, off);
    if (lane == 0) HH[row] = hh;
  }
}

// ---------------------------------------------------------------------------
// Softmax over N=256 logits per row; one wave per row, 8 values per lane.
// ---------------------------------------------------------------------------
__global__ __launch_bounds__(256) void softmax_kernel(const float* __restrict__ S,
                                                      bf16* __restrict__ Attn, int rows) {
  const int lane = threadIdx.x & 31;
  const int wave = threadIdx.x >> 5;
  const int row = blockIdx.x * 8 + wave;
  if (row >= rows) return;
  const float* s = S + (size_t)row * NNEUR;
  float v[8];
  float mx = -3.4e38f;
#pragma unroll
  for (int i = 0; i < 8; i++) { v[i] = s[lane + i * 32]; mx = fmaxf(mx, v[i]); }
#pragma unroll
  for (int off = 16; off > 0; off >>= 1) mx = fmaxf(mx, __shfl_xor(mx, off));
  float sum = 0.0f;
#pragma unroll
  for (int i = 0; i < 8; i++) { v[i] = expf(v[i] - mx); sum += v[i]; }
#pragma unroll
  for (int off = 16; off > 0; off >>= 1) sum += __shfl_xor(sum, off);
  const float inv = 1.0f / sum;
  bf16* a = Attn + (size_t)row * NNEUR;
#pragma unroll
  for (int i = 0; i < 8; i++) a[lane + i * 32] = (bf16)(v[i] * inv);
}

// ---------------------------------------------------------------------------
// Embedding: x[row,:] = wte[ids[row],:] + wpe[row % T,:]
// ---------------------------------------------------------------------------
__global__ __launch_bounds__(256) void embed_kernel(const int* __restrict__ ids,
                                                    const float* __restrict__ wte,
                                                    const float* __restrict__ wpe,
                                                    float* __restrict__ X) {
  const int row = blockIdx.x;
  const int t = row % NTOK;
  const int id = ids[row];
  for (int i = threadIdx.x; i < DMODEL; i += blockDim.x)
    X[(size_t)row * DMODEL + i] = wte[(size_t)id * DMODEL + i] + wpe[(size_t)t * DMODEL + i];
}

// f32 -> bf16, grid-stride
__global__ __launch_bounds__(256) void cvt_bf16_kernel(const float* __restrict__ in,
                                                       bf16* __restrict__ out, size_t n) {
  const size_t stride = (size_t)gridDim.x * blockDim.x;
  for (size_t i = (size_t)blockIdx.x * blockDim.x + threadIdx.x; i < n; i += stride)
    out[i] = (bf16)in[i];
}

// transpose + convert: in is (K x N) f32 row-major, out[n*K + k] = in[k*N + n] (bf16)
__global__ void tcvt_bf16_kernel(const float* __restrict__ in, bf16* __restrict__ out,
                                 int K, int N) {
  const int n = blockIdx.x * 16 + threadIdx.x;
  const int k = blockIdx.y * 16 + threadIdx.y;
  if (k < K && n < N) out[(size_t)n * K + k] = (bf16)in[(size_t)k * N + n];
}

// row squared-norms: PP[row] = sum_d P[row,d]^2  (one wave per row)
__global__ __launch_bounds__(256) void rownorm2_kernel(const float* __restrict__ P,
                                                       float* __restrict__ PP,
                                                       int rows, int D) {
  const int lane = threadIdx.x & 31;
  const int wave = threadIdx.x >> 5;
  const int row = blockIdx.x * 8 + wave;
  if (row >= rows) return;
  const float* p = P + (size_t)row * D;
  float s = 0.0f;
  for (int i = lane; i < D; i += 32) { const float v = p[i]; s += v * v; }
#pragma unroll
  for (int off = 16; off > 0; off >>= 1) s += __shfl_xor(s, off);
  if (lane == 0) PP[row] = s;
}

// ---------------------------------------------------------------------------
// host-side launch
// ---------------------------------------------------------------------------
static inline dim3 gemm_grid(int M, int N) {
  return dim3((unsigned)((N + 127) / 128), (unsigned)((M + 127) / 128), 1);
}

extern "C" void kernel_launch(void* const* d_in, const int* in_sizes, int n_in,
                              void* d_out, int out_size, void* d_ws, size_t ws_size,
                              hipStream_t stream) {
  (void)in_sizes; (void)n_in; (void)out_size; (void)ws_size;
  const int*   ids        = (const int*)d_in[0];
  const float* wte        = (const float*)d_in[1];
  const float* wpe        = (const float*)d_in[2];
  const float* ln1_w      = (const float*)d_in[3];
  const float* ln1_b      = (const float*)d_in[4];
  const float* positions  = (const float*)d_in[5];
  const float* interscale = (const float*)d_in[6];
  const float* vproj_w    = (const float*)d_in[7];
  const float* vproj_b    = (const float*)d_in[8];
  const float* oproj_w    = (const float*)d_in[9];
  const float* oproj_b    = (const float*)d_in[10];
  const float* ln2_w      = (const float*)d_in[11];
  const float* ln2_b      = (const float*)d_in[12];
  const float* up_w       = (const float*)d_in[13];
  const float* up_b       = (const float*)d_in[14];
  const float* down_w     = (const float*)d_in[15];
  const float* down_b     = (const float*)d_in[16];
  const float* lnf_w      = (const float*)d_in[17];
  const float* lnf_b      = (const float*)d_in[18];
  float* out = (float*)d_out;

  // workspace carve-up (deterministic, recomputed every call)
  char* wsb = (char*)d_ws;
  size_t off = 0;
  auto alloc = [&](size_t bytes) -> void* {
    void* p = wsb + off;
    off = (off + bytes + 255) & ~(size_t)255;
    return p;
  };
  float* x      = (float*)alloc((size_t)NROWS * DMODEL * 4);
  bf16*  h      = (bf16*) alloc((size_t)NROWS * DMODEL * 2);
  float* hh     = (float*)alloc((size_t)NROWS * 4);
  float* S      = (float*)alloc((size_t)NROWS * NNEUR * 4);
  bf16*  attn   = (bf16*) alloc((size_t)NROWS * NNEUR * 2);
  bf16*  tmp    = (bf16*) alloc((size_t)NROWS * DMODEL * 2);
  bf16*  u      = (bf16*) alloc((size_t)NROWS * NHID * 2);
  bf16*  vt     = (bf16*) alloc((size_t)DMODEL * NNEUR * 2);
  bf16*  wte_bf = (bf16*) alloc((size_t)VOCAB * DMODEL * 2);
  bf16*  pos_bf = (bf16*) alloc((size_t)NLAYER * NNEUR * DMODEL * 2);
  float* pp     = (float*)alloc((size_t)NLAYER * NNEUR * 4);
  bf16*  vprojT = (bf16*) alloc((size_t)NLAYER * DMODEL * DMODEL * 2);
  bf16*  oprojT = (bf16*) alloc((size_t)NLAYER * DMODEL * DMODEL * 2);
  bf16*  upT    = (bf16*) alloc((size_t)NLAYER * NHID * DMODEL * 2);
  bf16*  downT  = (bf16*) alloc((size_t)NLAYER * DMODEL * NHID * 2);

  // ---- weight prep (bf16 convert + transpose; cheap, L2-resident afterwards)
  cvt_bf16_kernel<<<2048, 256, 0, stream>>>(wte, wte_bf, (size_t)VOCAB * DMODEL);
  cvt_bf16_kernel<<<512, 256, 0, stream>>>(positions, pos_bf,
                                           (size_t)NLAYER * NNEUR * DMODEL);
  rownorm2_kernel<<<(NLAYER * NNEUR) / 8, 256, 0, stream>>>(positions, pp,
                                                            NLAYER * NNEUR, DMODEL);
  {
    dim3 blk(16, 16);
    dim3 gDD((DMODEL + 15) / 16, (DMODEL + 15) / 16);
    dim3 gDH((NHID + 15) / 16, (DMODEL + 15) / 16);   // in (D x H) -> out (H x D)
    dim3 gHD((DMODEL + 15) / 16, (NHID + 15) / 16);   // in (H x D) -> out (D x H)
    for (int l = 0; l < NLAYER; l++) {
      tcvt_bf16_kernel<<<gDD, blk, 0, stream>>>(vproj_w + (size_t)l * DMODEL * DMODEL,
                                                vprojT + (size_t)l * DMODEL * DMODEL,
                                                DMODEL, DMODEL);
      tcvt_bf16_kernel<<<gDD, blk, 0, stream>>>(oproj_w + (size_t)l * DMODEL * DMODEL,
                                                oprojT + (size_t)l * DMODEL * DMODEL,
                                                DMODEL, DMODEL);
      tcvt_bf16_kernel<<<gDH, blk, 0, stream>>>(up_w + (size_t)l * DMODEL * NHID,
                                                upT + (size_t)l * NHID * DMODEL,
                                                DMODEL, NHID);
      tcvt_bf16_kernel<<<gHD, blk, 0, stream>>>(down_w + (size_t)l * NHID * DMODEL,
                                                downT + (size_t)l * DMODEL * NHID,
                                                NHID, DMODEL);
    }
  }

  // ---- embedding
  embed_kernel<<<NROWS, 256, 0, stream>>>(ids, wte, wpe, x);

  // ---- transformer layers
  for (int l = 0; l < NLAYER; l++) {
    const bf16* pl   = pos_bf + (size_t)l * NNEUR * DMODEL;
    const float* ppl = pp + (size_t)l * NNEUR;

    // h = LN1(x), hh = ||h||^2 per row
    ln_kernel<<<NROWS / 8, 256, 0, stream>>>(x, ln1_w + (size_t)l * DMODEL,
                                             ln1_b + (size_t)l * DMODEL, h, hh,
                                             NROWS, DMODEL);
    // scores S = scale / (||h-p|| + 0.1)   (DIST epilogue on h @ p^T)
    gemm_bf16_kernel<<<gemm_grid(NROWS, NNEUR), 256, 0, stream>>>(
        h, pl, S, nullptr, nullptr, hh, ppl, interscale + (size_t)l * NNEUR,
        NROWS, NNEUR, DMODEL, EPI_DIST);
    // attn = softmax(S)
    softmax_kernel<<<NROWS / 8, 256, 0, stream>>>(S, attn, NROWS);
    // values^T = (p @ vproj_w + vproj_b)^T  -> vt[d*NNEUR + n]
    gemm_bf16_kernel<<<gemm_grid(NNEUR, DMODEL), 256, 0, stream>>>(
        pl, vprojT + (size_t)l * DMODEL * DMODEL, nullptr, vt,
        vproj_b + (size_t)l * DMODEL, nullptr, nullptr, nullptr,
        NNEUR, DMODEL, DMODEL, EPI_BF16_T);
    // tmp = attn @ values
    gemm_bf16_kernel<<<gemm_grid(NROWS, DMODEL), 256, 0, stream>>>(
        attn, vt, nullptr, tmp, nullptr, nullptr, nullptr, nullptr,
        NROWS, DMODEL, NNEUR, EPI_BF16);
    // x += tmp @ oproj_w + oproj_b
    gemm_bf16_kernel<<<gemm_grid(NROWS, DMODEL), 256, 0, stream>>>(
        tmp, oprojT + (size_t)l * DMODEL * DMODEL, x, nullptr,
        oproj_b + (size_t)l * DMODEL, nullptr, nullptr, nullptr,
        NROWS, DMODEL, DMODEL, EPI_RES);
    // h = LN2(x)
    ln_kernel<<<NROWS / 8, 256, 0, stream>>>(x, ln2_w + (size_t)l * DMODEL,
                                             ln2_b + (size_t)l * DMODEL, h, nullptr,
                                             NROWS, DMODEL);
    // u = gelu(h @ up_w + up_b)
    gemm_bf16_kernel<<<gemm_grid(NROWS, NHID), 256, 0, stream>>>(
        h, upT + (size_t)l * NHID * DMODEL, nullptr, u,
        up_b + (size_t)l * NHID, nullptr, nullptr, nullptr,
        NROWS, NHID, DMODEL, EPI_GELU);
    // x += u @ down_w + down_b
    gemm_bf16_kernel<<<gemm_grid(NROWS, DMODEL), 256, 0, stream>>>(
        u, downT + (size_t)l * DMODEL * NHID, x, nullptr,
        down_b + (size_t)l * DMODEL, nullptr, nullptr, nullptr,
        NROWS, DMODEL, NHID, EPI_RES);
  }

  // ---- final LN + tied lm_head: out = LN(x) @ wte^T  (B=wte^T is K-contiguous in wte)
  ln_kernel<<<NROWS / 8, 256, 0, stream>>>(x, lnf_w, lnf_b, h, nullptr, NROWS, DMODEL);
  gemm_bf16_kernel<<<gemm_grid(NROWS, VOCAB), 256, 0, stream>>>(
      h, wte_bf, out, nullptr, nullptr, nullptr, nullptr, nullptr,
      NROWS, VOCAB, DMODEL, EPI_F32);
}
